// MutiHeadattention_7593502179569
// MI455X (gfx1250) — matrix operand order repaired and checked by reading, not compiled, not run on hardware
//
#include <hip/hip_runtime.h>
#include <hip/hip_bf16.h>
#include <math.h>

// Problem dims (fixed by the reference)
#define BB    4
#define SS    2048
#define HH    16
#define DD    64
#define HID   1024
#define MROWS (BB * SS)   // 8192

typedef __attribute__((ext_vector_type(16))) _Float16 v16h;
typedef __attribute__((ext_vector_type(8)))  _Float16 v8h;
typedef __attribute__((ext_vector_type(8)))  float    v8f;
typedef __attribute__((ext_vector_type(4)))  unsigned int v4u;
typedef __attribute__((ext_vector_type(8)))  int      v8i;
typedef __attribute__((ext_vector_type(4)))  int      v4i;

// ---------------------------------------------------------------------------
// Fragment loaders (wave32 WMMA layouts per CDNA5 ISA 7.12.2 / 7.12.5)
//
// A fragment, 16x32 f16, element (m, k) at base[m*stride + k]:
//   lane L: m = L%16, g = L/16
//   halfs[0..7]  = K = g*8 + 0..7        (16B contiguous)
//   halfs[8..15] = K = 16 + g*8 + 0..7   (16B contiguous)
// ---------------------------------------------------------------------------
__device__ inline v16h load_fragA(const _Float16* base, int stride) {
  int lane = threadIdx.x & 31;
  int m = lane & 15, g = lane >> 4;
  const _Float16* p = base + (size_t)m * stride + g * 8;
  v8h lo = *(const v8h*)(p);
  v8h hi = *(const v8h*)(p + 16);
  return __builtin_shufflevector(lo, hi, 0,1,2,3,4,5,6,7,8,9,10,11,12,13,14,15);
}

// ---------------------------------------------------------------------------
// B fragment, 32x16 f16, element (k, n) at base[n*stride + k] (N-major memory):
//   lane L: n = L%16, g = L/16
//   halfs[0..7]  = K = g*8 + 0..7        (16B contiguous)
//   halfs[8..15] = K = 16 + g*8 + 0..7   (16B contiguous)
// ---------------------------------------------------------------------------
__device__ inline v16h load_fragB(const _Float16* base, int stride) {
  int lane = threadIdx.x & 31;
  int nn = lane & 15, g = lane >> 4;
  const _Float16* p = base + (size_t)nn * stride + g * 8;
  v8h lo = *(const v8h*)(p);
  v8h hi = *(const v8h*)(p + 16);
  return __builtin_shufflevector(lo, hi, 0,1,2,3,4,5,6,7,8,9,10,11,12,13,14,15);
}

__device__ inline v8f wmma16(v16h a, v16h b, v8f c) {
  // D = A(16x32) * B(32x16) + C, f32 accumulate
  return __builtin_amdgcn_wmma_f32_16x16x32_f16(false, a, false, b, (short)0, c,
                                                false, false);
}

// ---------------------------------------------------------------------------
// Tensor Data Mover: async 2D tile load (global -> LDS), f16 elements.
// D# per CDNA5 ISA ch.8: group0 = {flags, lds_addr, global_addr, type=2},
// group1 = {mask=0, data_size=1(2B), tensor dims, tile dims, dim0 stride}.
// Tracked with TENSORcnt; not in a cluster so workgroup_mask must be 0.
// ---------------------------------------------------------------------------
__device__ inline void tdm_load_2d_f16(unsigned lds_off, const void* gsrc,
                                       unsigned tile_d0, unsigned tile_d1,
                                       unsigned stride_elems) {
  unsigned long long ga = (unsigned long long)gsrc;
  v4u g0;
  g0.x = 1u;                                         // count=1, user-mode load
  g0.y = lds_off;                                    // [63:32]  lds_addr (bytes)
  g0.z = (unsigned)(ga & 0xffffffffu);               // [95:64]  global_addr lo
  g0.w = (unsigned)((ga >> 32) & 0x01ffffffu)        // [120:96] global_addr hi
         | (2u << 30);                               // [127:126] type = 2
  v8i g1;
  g1[0] = (int)(1u << 16);                           // data_size=1 (2 bytes)
  g1[1] = (int)((tile_d0 & 0xffffu) << 16);          // tensor_dim0[15:0]
  g1[2] = (int)((tile_d1 & 0xffffu) << 16);          // tensor_dim0 hi=0, dim1 lo
  g1[3] = (int)((tile_d0 & 0xffffu) << 16);          // dim1 hi=0, tile_dim0
  g1[4] = (int)(tile_d1 & 0xffffu);                  // tile_dim1 (tile_dim2=0)
  g1[5] = (int)stride_elems;                         // tensor_dim0_stride lo32
  g1[6] = 0;
  g1[7] = 0;
  v4i z4 = {0, 0, 0, 0};
#if __clang_major__ >= 23
  v8i z8 = {0, 0, 0, 0, 0, 0, 0, 0};
  __builtin_amdgcn_tensor_load_to_lds(g0, g1, z4, z4, z8, 0);
#else
  __builtin_amdgcn_tensor_load_to_lds(g0, g1, z4, z4, 0);
#endif
}

__device__ inline unsigned lds_offset_of(const void* p) {
  // Generic pointers into the LDS aperture carry the LDS byte offset in the
  // low 32 bits (CDNA5 ISA 10.2).
  return (unsigned)(unsigned long long)p;
}

// ---------------------------------------------------------------------------
// Prep kernels: f32 -> f16 convert, and [K x N] f32 -> [N x K] f16 transpose
// ---------------------------------------------------------------------------
__global__ void cvt_f32_f16_kernel(const float* __restrict__ in,
                                   _Float16* __restrict__ out, int nel) {
  int i = blockIdx.x * blockDim.x + threadIdx.x;
  if (i < nel) out[i] = (_Float16)in[i];
}

__global__ void transpose_w_kernel(const float* __restrict__ in,
                                   _Float16* __restrict__ out, int K, int N) {
  int i = blockIdx.x * blockDim.x + threadIdx.x;
  if (i < K * N) {
    int k = i / N, nn = i - k * N;
    out[(size_t)nn * K + k] = (_Float16)in[i];
  }
}

// ---------------------------------------------------------------------------
// QKV projection: out = (xh[8192x64] @ W[64x1024] + b) * scale
//   mode 0/1: store [bh][s][d] f16 (Q pre-scaled by 1/sqrt(D), K unscaled)
//   mode 2:   store V transposed [bh][d][s] f16 (for contiguous PV B-frags)
// Wave = 16 rows x 64 cols. grid = (8192/16, 16/8) = (512, 2), block = 256.
// ---------------------------------------------------------------------------
__global__ __launch_bounds__(256) void qkv_proj_kernel(
    const _Float16* __restrict__ xh, const _Float16* __restrict__ wt,
    const float* __restrict__ bias, _Float16* __restrict__ out,
    int mode, float scale) {
  int w = threadIdx.x >> 5;
  int lane = threadIdx.x & 31;
  int n = lane & 15, g = lane >> 4;
  int m0 = blockIdx.x * 16;
  int nc0 = (blockIdx.y * 8 + w) * 64;
  int b_ = m0 / SS;              // 16-row tile never crosses a batch boundary
  int s0 = m0 - b_ * SS;

  v16h a0 = load_fragA(xh + (size_t)m0 * DD, DD);        // K = 0..31
  v16h a1 = load_fragA(xh + (size_t)m0 * DD + 32, DD);   // K = 32..63

#pragma unroll
  for (int t = 0; t < 4; ++t) {
    int ncol = nc0 + t * 16;
    v16h b0 = load_fragB(wt + (size_t)ncol * DD, DD);
    v16h b1 = load_fragB(wt + (size_t)ncol * DD + 32, DD);
    v8f c = {};
    c = wmma16(a0, b0, c);
    c = wmma16(a1, b1, c);
    float bv = bias[ncol + n];
    int h = (ncol + n) >> 6;
    int d = (ncol + n) & 63;
    int bh = b_ * HH + h;
#pragma unroll
    for (int r = 0; r < 8; ++r) {
      int s = s0 + r + 8 * g;
      float v = (c[r] + bv) * scale;
      if (mode == 2)
        out[(size_t)bh * DD * SS + (size_t)d * SS + s] = (_Float16)v;
      else
        out[(size_t)bh * SS * DD + (size_t)s * DD + d] = (_Float16)v;
    }
  }
}

// ---------------------------------------------------------------------------
// Causal flash attention. One wave per 16-row q-tile; all 8 waves of a block
// share one (b,h). K/V key-blocks (32 keys) are staged into LDS once per
// block by the Tensor Data Mover, double-buffered (TDM for j+32 overlaps
// compute on j). grid = (S/16/8, B*H) = (16, 64), block = 256, LDS = 24KB.
// ---------------------------------------------------------------------------
__global__ __launch_bounds__(256) void attn_kernel(
    const _Float16* __restrict__ qh, const _Float16* __restrict__ kh,
    const _Float16* __restrict__ vt, _Float16* __restrict__ ao) {
  __shared__ __align__(128) _Float16 kstage[2][32 * DD];   // [key][d]   4KB x2
  __shared__ __align__(128) _Float16 vstage[2][DD * 32];   // [d][key]   4KB x2
  __shared__ __align__(128) _Float16 pstage[8][16 * 32];   // per-wave P 8KB

  int w = threadIdx.x >> 5;
  int lane = threadIdx.x & 31;
  int n = lane & 15, g = lane >> 4;
  int qtile = blockIdx.x * 8 + w;          // 0..127
  int bh = blockIdx.y;                     // 0..63
  int b = bh >> 4, h = bh & 15;
  int qi0 = qtile * 16;

  const _Float16* qbase = qh + (size_t)bh * SS * DD + (size_t)qi0 * DD;
  const _Float16* kbase = kh + (size_t)bh * SS * DD;
  const _Float16* vbase = vt + (size_t)bh * DD * SS;

  v16h aq0 = load_fragA(qbase, DD);        // d = 0..31 (Q pre-scaled by 1/8)
  v16h aq1 = load_fragA(qbase + 32, DD);   // d = 32..63

  v8f acc[4] = {};
  float mrow[8], lrow[8];
#pragma unroll
  for (int r = 0; r < 8; ++r) { mrow[r] = -3.0e38f; lrow[r] = 0.0f; }

  const int myqend = qi0 + 15;
  const int qendBlk = (blockIdx.x * 8 + 7) * 16 + 15;   // block max diagonal

  // Prologue: TDM-stage key block j=0 into buffer 0 (wave 0 issues).
  if (w == 0) {
    tdm_load_2d_f16(lds_offset_of(&kstage[0][0]), kbase, DD, 32, DD);
    tdm_load_2d_f16(lds_offset_of(&vstage[0][0]), vbase, 32, DD, SS);
  }

  int p = 0;
  for (int j = 0; j <= qendBlk; j += 32) {
    if (w == 0) __builtin_amdgcn_s_wait_tensorcnt(0);   // buf p staged
    __syncthreads();
    // Prefetch next key block into the other buffer while we compute.
    if (w == 0 && j + 32 <= qendBlk) {
      tdm_load_2d_f16(lds_offset_of(&kstage[p ^ 1][0]),
                      kbase + (size_t)(j + 32) * DD, DD, 32, DD);
      tdm_load_2d_f16(lds_offset_of(&vstage[p ^ 1][0]),
                      vbase + (j + 32), 32, DD, SS);
    }

    if (j <= myqend) {   // wave-uniform: EXEC stays all-ones inside
      // ---- scores: S = Q @ K^T for keys [j, j+32) from LDS ----
      v16h bk00 = load_fragB(&kstage[p][0], DD);            // keys 0..15, d 0..31
      v16h bk01 = load_fragB(&kstage[p][32], DD);           //             d 32..63
      v16h bk10 = load_fragB(&kstage[p][16 * DD], DD);      // keys 16..31
      v16h bk11 = load_fragB(&kstage[p][16 * DD + 32], DD);
      v8f c0 = {}, c1 = {};
      c0 = wmma16(aq0, bk00, c0);
      c0 = wmma16(aq1, bk01, c0);
      c1 = wmma16(aq0, bk10, c1);
      c1 = wmma16(aq1, bk11, c1);

      // ---- causal mask + online softmax (row stats via 16-lane shuffles) ----
#pragma unroll
      for (int r = 0; r < 8; ++r) {
        int m = r + 8 * g;
        int qrow = qi0 + m;
        float s0 = (j + n > qrow)      ? -3.0e38f : c0[r];
        float s1 = (j + 16 + n > qrow) ? -3.0e38f : c1[r];
        float bm = fmaxf(s0, s1);
        bm = fmaxf(bm, __shfl_xor(bm, 1, 16));
        bm = fmaxf(bm, __shfl_xor(bm, 2, 16));
        bm = fmaxf(bm, __shfl_xor(bm, 4, 16));
        bm = fmaxf(bm, __shfl_xor(bm, 8, 16));
        float mnew = fmaxf(mrow[r], bm);
        float sc = __expf(mrow[r] - mnew);
        float p0 = __expf(s0 - mnew);
        float p1 = __expf(s1 - mnew);
        float rs = p0 + p1;
        rs += __shfl_xor(rs, 1, 16);
        rs += __shfl_xor(rs, 2, 16);
        rs += __shfl_xor(rs, 4, 16);
        rs += __shfl_xor(rs, 8, 16);
        lrow[r] = lrow[r] * sc + rs;
        mrow[r] = mnew;
        acc[0][r] *= sc; acc[1][r] *= sc; acc[2][r] *= sc; acc[3][r] *= sc;
        // stage P (C layout -> row-major 16x32 in LDS for A-frag reload)
        pstage[w][m * 32 + n]      = (_Float16)p0;
        pstage[w][m * 32 + 16 + n] = (_Float16)p1;
      }
      __asm__ volatile("" ::: "memory");  // keep DS store->load program order

      // ---- O += P @ V  (A from LDS, B from LDS V tile, K = 32 keys) ----
      v16h ap = load_fragA(&pstage[w][0], 32);
#pragma unroll
      for (int t = 0; t < 4; ++t) {
        v16h bv = load_fragB(&vstage[p][t * 16 * 32], 32);
        acc[t] = wmma16(ap, bv, acc[t]);
      }
    }

    __syncthreads();   // compute done before buf p is overwritten next round
    p ^= 1;
  }

  // ---- normalize and store to attn_out [B*S][HID] f16 ----
  int row0 = b * SS + qi0;
#pragma unroll
  for (int r = 0; r < 8; ++r) {
    int m = r + 8 * g;
    float inv = 1.0f / lrow[r];
    size_t o = (size_t)(row0 + m) * HID + h * DD + n;
#pragma unroll
    for (int t = 0; t < 4; ++t)
      ao[o + t * 16] = (_Float16)(acc[t][r] * inv);
  }
}

// ---------------------------------------------------------------------------
// Output projection: out[8192x64] = ao[8192x1024] @ Wo[1024x64] + bo  (f32 out)
// Wave = 16 rows x 64 cols, 32 k-steps. grid = 512/8 = 64 blocks of 256.
// ---------------------------------------------------------------------------
__global__ __launch_bounds__(256) void out_proj_kernel(
    const _Float16* __restrict__ ao, const _Float16* __restrict__ wto,
    const float* __restrict__ bo, float* __restrict__ out) {
  int w = threadIdx.x >> 5;
  int lane = threadIdx.x & 31;
  int n = lane & 15, g = lane >> 4;
  int m0 = (blockIdx.x * 8 + w) * 16;

  v8f acc[4] = {};
  for (int kk = 0; kk < HID; kk += 32) {
    v16h a = load_fragA(ao + (size_t)m0 * HID + kk, HID);
#pragma unroll
    for (int t = 0; t < 4; ++t) {
      v16h bfr = load_fragB(wto + (size_t)(t * 16) * HID + kk, HID);
      acc[t] = wmma16(a, bfr, acc[t]);
    }
  }
#pragma unroll
  for (int t = 0; t < 4; ++t) {
    float bias = bo[t * 16 + n];
#pragma unroll
    for (int r = 0; r < 8; ++r) {
      int row = m0 + r + 8 * g;
      out[(size_t)row * DD + t * 16 + n] = acc[t][r] + bias;
    }
  }
}

// ---------------------------------------------------------------------------
// Host launcher
// ---------------------------------------------------------------------------
extern "C" void kernel_launch(void* const* d_in, const int* in_sizes, int n_in,
                              void* d_out, int out_size, void* d_ws, size_t ws_size,
                              hipStream_t stream) {
  const float* x  = (const float*)d_in[0];
  const float* Wq = (const float*)d_in[1];
  const float* bq = (const float*)d_in[2];
  const float* Wk = (const float*)d_in[3];
  const float* bk = (const float*)d_in[4];
  const float* Wv = (const float*)d_in[5];
  const float* bv = (const float*)d_in[6];
  const float* Wo = (const float*)d_in[7];
  const float* bo = (const float*)d_in[8];
  float* out = (float*)d_out;

  // Workspace layout (f16 staging buffers), total ~65.5 MB
  char* ws = (char*)d_ws;
  _Float16* xh  = (_Float16*)(ws);                         // 8192*64      = 1 MB
  _Float16* wtq = (_Float16*)(ws + (1u << 20));            // [1024][64]   = 128 KB
  _Float16* wtk = wtq + HID * DD;
  _Float16* wtv = wtk + HID * DD;
  _Float16* wto = wtv + HID * DD;                          // Wo^T [64][1024]
  _Float16* qh  = (_Float16*)(ws + (1u << 20) + 4u * (HID * DD * 2)); // 16 MB each
  _Float16* kh  = qh + (size_t)BB * HH * SS * DD;
  _Float16* vt  = kh + (size_t)BB * HH * SS * DD;
  _Float16* ao  = vt + (size_t)BB * HH * SS * DD;

  const float inv_sqrt_d = 0.125f;  // 1/sqrt(64)

  // Stage 0: convert x, transpose weights to N-major f16
  {
    int nel = MROWS * DD;
    cvt_f32_f16_kernel<<<(nel + 255) / 256, 256, 0, stream>>>(x, xh, nel);
    int nw = DD * HID;
    transpose_w_kernel<<<(nw + 255) / 256, 256, 0, stream>>>(Wq, wtq, DD, HID);
    transpose_w_kernel<<<(nw + 255) / 256, 256, 0, stream>>>(Wk, wtk, DD, HID);
    transpose_w_kernel<<<(nw + 255) / 256, 256, 0, stream>>>(Wv, wtv, DD, HID);
    transpose_w_kernel<<<(nw + 255) / 256, 256, 0, stream>>>(Wo, wto, HID, DD);
  }

  // Stage 1: QKV projections (WMMA). Q pre-scaled by 1/sqrt(D).
  {
    dim3 grid(MROWS / 16, 2);
    qkv_proj_kernel<<<grid, 256, 0, stream>>>(xh, wtq, bq, qh, 0, inv_sqrt_d);
    qkv_proj_kernel<<<grid, 256, 0, stream>>>(xh, wtk, bk, kh, 1, 1.0f);
    qkv_proj_kernel<<<grid, 256, 0, stream>>>(xh, wtv, bv, vt, 2, 1.0f);
  }

  // Stage 2: causal flash attention (TDM-staged K/V + WMMA + online softmax)
  {
    dim3 grid(SS / 16 / 8, BB * HH);
    attn_kernel<<<grid, 256, 0, stream>>>(qh, kh, vt, ao);
  }

  // Stage 3: output projection (WMMA) + bias, f32 store
  {
    out_proj_kernel<<<MROWS / 16 / 8, 256, 0, stream>>>(ao, wto, bo, out);
  }
}
